// MultiHeadAttention_36129264894007
// MI455X (gfx1250) — compile-verified, matrix operand order checked by reading, not run on hardware
//
#include <hip/hip_runtime.h>
#include <hip/hip_bf16.h>

// ---------------- problem constants ----------------
#define BATCH   2
#define LQ      2048
#define LK      2048
#define DQRY    256
#define DMODEL  512
#define NHEAD   8
#define DK      64
#define DV      64

typedef __attribute__((ext_vector_type(16))) _Float16 v16h;
typedef __attribute__((ext_vector_type(8)))  _Float16 h8;
typedef __attribute__((ext_vector_type(8)))  float    v8f;
typedef __attribute__((ext_vector_type(4)))  float    f4v;
typedef unsigned int u32x4 __attribute__((ext_vector_type(4)));
typedef int          i32x8 __attribute__((ext_vector_type(8)));
typedef int          i32x4 __attribute__((ext_vector_type(4)));

// D = A*B + C, 16x16x32 f16 -> f32
__device__ __forceinline__ v8f wmma16(v16h a, v16h b, v8f c) {
    return __builtin_amdgcn_wmma_f32_16x16x32_f16(false, a, false, b, (short)0, c,
                                                  false, false);
}

// A/B fragment loader from f16 row (row-major, contiguous K).
// CDNA5 16-bit A layout: lanes 0-15 hold K = kb..kb+7 (VGPR0-3) and
// kb+16..kb+23 (VGPR4-7), lanes 16-31 the +8 shifted groups (kb=(lane>>4)*8).
__device__ __forceinline__ v16h load_frag_f16(const _Float16* rowp, int kb) {
    h8 lo = *(const h8*)(rowp + kb);
    h8 hi = *(const h8*)(rowp + kb + 16);
    v16h r;
#pragma unroll
    for (int i = 0; i < 8; ++i) { r[i] = lo[i]; r[8 + i] = hi[i]; }
    return r;
}

// Same fragment but converting from f32 (global or LDS) with optional scale.
__device__ __forceinline__ v16h load_frag_f32(const float* rowp, int kb, float s) {
    f4v a0 = *(const f4v*)(rowp + kb);
    f4v a1 = *(const f4v*)(rowp + kb + 4);
    f4v a2 = *(const f4v*)(rowp + kb + 16);
    f4v a3 = *(const f4v*)(rowp + kb + 20);
    v16h r;
#pragma unroll
    for (int i = 0; i < 4; ++i) {
        r[i]      = (_Float16)(a0[i] * s);
        r[4 + i]  = (_Float16)(a1[i] * s);
        r[8 + i]  = (_Float16)(a2[i] * s);
        r[12 + i] = (_Float16)(a3[i] * s);
    }
    return r;
}

// Tensor Data Mover: DMA a 16x64 f16 tile (contiguous rows, stride 64) from
// global memory into LDS at byte offset lds_off.  D# built per ISA 08_async_tensor:
// group0 = {count=1 | lds_addr | global_addr | type=2}, group1 = dims/strides.
__device__ __forceinline__ void tdm_load_k_tile(const _Float16* gsrc,
                                                unsigned lds_off, int rows) {
    unsigned long long ga = (unsigned long long)(size_t)gsrc;
    u32x4 g0;
    g0[0] = 1u;                                          // count=1 (valid user D#)
    g0[1] = lds_off;                                     // lds_addr (bytes)
    g0[2] = (unsigned)(ga & 0xffffffffu);                // global_addr[31:0]
    g0[3] = (unsigned)((ga >> 32) & 0x01ffffffu) | 0x80000000u;  // addr[56:32]|type=2

    i32x8 g1;
    g1[0] = 0x00010000;            // workgroup_mask=0, data_size=1 (2 bytes)
    g1[1] = (int)(64u << 16);      // tensor_dim0 = 64   (bits [79:48], low half)
    g1[2] = (int)((unsigned)rows << 16);  // tensor_dim1 = rows (bits [111:80])
    g1[3] = (int)(64u << 16);      // tile_dim0 = 64     (bits [127:112])
    g1[4] = rows;                  // tile_dim1 = rows   (bits [143:128])
    g1[5] = 64;                    // tensor_dim0_stride = 64 (bits [207:160])
    g1[6] = 0;
    g1[7] = 0;

    i32x4 z4 = {0, 0, 0, 0};       // groups 2/3 unused (2D tensor)
#if defined(__clang_major__) && (__clang_major__ >= 23)
    i32x8 z8 = {0, 0, 0, 0, 0, 0, 0, 0};
    __builtin_amdgcn_tensor_load_to_lds(g0, g1, z4, z4, z8, 0);
#else
    __builtin_amdgcn_tensor_load_to_lds(g0, g1, z4, z4, 0);
#endif
}

// ---------------- K0: weight transpose + f32->f16 ----------------
__global__ void wtrans_kernel(const float* __restrict__ W, _Float16* __restrict__ WT,
                              int R, int C) {
    int idx = blockIdx.x * 256 + threadIdx.x;
    if (idx < R * C) {
        int r = idx / C, c = idx % C;
        WT[(size_t)c * R + r] = (_Float16)W[idx];
    }
}

// ---------------- K1: input projections ----------------
// mode 0: qh[b][h][lq][dk] (scaled 1/8)   mode 1: kh[b][h][lk][dk]
// mode 2: vhT[b][h][dk][lk]               X: [4096][Din] f32, WT: [512][Din] f16
__global__ __launch_bounds__(256) void proj_kernel(const float* __restrict__ X,
                                                   const _Float16* __restrict__ WT,
                                                   _Float16* __restrict__ OUT,
                                                   int Din, int mode) {
    __shared__ float tile[8][256];  // one 16x16 f32 tile per wave
    const int lane = threadIdx.x & 31;
    const int wave = threadIdx.x >> 5;
    const int rb   = blockIdx.x;                    // 16-row block (0..255)
    const int n0   = blockIdx.y * 128 + wave * 16;  // output-col tile base
    const int mrow = lane & 15;
    const int kb   = (lane >> 4) * 8;

    const float* xrow = X + (size_t)(rb * 16 + mrow) * Din;
    const _Float16* wrow = WT + (size_t)(n0 + mrow) * Din;

    v8f acc = {};
    for (int kk = 0; kk < Din; kk += 32) {
        v16h a = load_frag_f32(xrow + kk, kb, 1.0f);
        v16h b = load_frag_f16(wrow + kk, kb);
        acc = wmma16(a, b, acc);
    }

    // stage tile in LDS; D-frag lane holds (m = kb+v, n = mrow)
    if (mode == 2) {  // transposed staging: tile[n][m] -> rows become d-index
        f4v s0 = {acc[0], acc[1], acc[2], acc[3]};
        f4v s1 = {acc[4], acc[5], acc[6], acc[7]};
        *(f4v*)&tile[wave][mrow * 16 + kb]     = s0;
        *(f4v*)&tile[wave][mrow * 16 + kb + 4] = s1;
    } else {
#pragma unroll
        for (int v = 0; v < 8; ++v) tile[wave][(kb + v) * 16 + mrow] = acc[v];
    }
    __syncthreads();

    if (lane < 16) {
        const float scale = (mode == 0) ? 0.125f : 1.0f;  // 1/sqrt(64) folded into qh
        const int h  = n0 >> 6;
        const int d0 = n0 & 63;
        const int bb  = (rb * 16) >> 11;
        const int l0  = (rb * 16) & 2047;
        const float* trow = &tile[wave][lane * 16];
        h8 o0, o1;
#pragma unroll
        for (int i = 0; i < 8; ++i) {
            o0[i] = (_Float16)(trow[i] * scale);
            o1[i] = (_Float16)(trow[8 + i] * scale);
        }
        _Float16* dst;
        if (mode == 2)  // row = d, cols = 16 consecutive Lk
            dst = OUT + (((size_t)(bb * NHEAD + h) * DK) + d0 + lane) * LK + l0;
        else            // row = L, cols = 16 consecutive d
            dst = OUT + (((size_t)(bb * NHEAD + h) * LQ) + l0 + lane) * DK + d0;
        *(h8*)dst       = o0;
        *(h8*)(dst + 8) = o1;
    }
}

// ---------------- K2: scores + mask + softmax + attn-out + P*V ----------------
// one WG = (b, h, 16 q-rows); full 16x2048 score strip lives in LDS.
// K-tiles are staged into LDS by the Tensor Data Mover (double-buffered per wave).
__global__ __launch_bounds__(256) void attn_kernel(const _Float16* __restrict__ qh,
                                                   const _Float16* __restrict__ kh,
                                                   const _Float16* __restrict__ vhT,
                                                   const int* __restrict__ mask,
                                                   float* __restrict__ attnOut,
                                                   _Float16* __restrict__ ctx) {
    extern __shared__ float smem[];
    float* sc  = smem;               // [16][2048] scores / probabilities (128KB)
    float* red = smem + 16 * LK;     // [4][16][16] PV partial reduction (4KB)
    _Float16* kbuf = (_Float16*)(smem + 16 * LK + 1024);  // 8 waves x 2 x 16x64 f16
    const unsigned kbuf_base = (unsigned)((16 * LK + 1024) * 4);  // byte off in LDS

    const int lane = threadIdx.x & 31;
    const int wave = threadIdx.x >> 5;
    const int lq0  = blockIdx.x * 16;
    const int hh   = blockIdx.y;
    const int bb   = blockIdx.z;
    const int mrow = lane & 15;
    const int kb   = (lane >> 4) * 8;

    // ---- phase 1: S = (q/sqrt(dk)) K^T, strip of 16 x 2048 ----
    const _Float16* qbase = qh + (((size_t)bb * NHEAD + hh) * LQ + lq0 + mrow) * DK;
    v16h a0 = load_frag_f16(qbase, kb);        // K-dim 0..31
    v16h a1 = load_frag_f16(qbase + 32, kb);   // K-dim 32..63
    const _Float16* khead = kh + ((size_t)bb * NHEAD + hh) * LK * DK;

    const unsigned myk_off = kbuf_base + (unsigned)wave * 4096u;  // 2 x 2KB slots
    _Float16* mykt = kbuf + wave * 2048;

    // prologue: TDM tile 0 into slot 0
    tdm_load_k_tile(khead + (size_t)(wave * 256) * DK, myk_off, 16);

    for (int t = 0; t < 16; ++t) {
        const int n0  = wave * 256 + t * 16;
        const int buf = t & 1;
        if (t < 15) {  // TDM next tile into the other slot while we compute
            tdm_load_k_tile(khead + (size_t)(n0 + 16) * DK,
                            myk_off + (unsigned)((t + 1) & 1) * 2048u, 16);
            __builtin_amdgcn_s_wait_tensorcnt(1);   // tile t landed
        } else {
            __builtin_amdgcn_s_wait_tensorcnt(0);
        }
        const _Float16* krow = mykt + buf * 1024 + mrow * DK;  // LDS tile row
        v8f s = {};
        s = wmma16(a0, load_frag_f16(krow, kb), s);
        s = wmma16(a1, load_frag_f16(krow + 32, kb), s);
#pragma unroll
        for (int v = 0; v < 8; ++v) sc[(size_t)(kb + v) * LK + n0 + mrow] = s[v];
    }
    __syncthreads();

    // ---- phase 2: mask + softmax over each row (16 threads per row) ----
    {
        const int r = threadIdx.x >> 4;   // row in strip
        const int j = threadIdx.x & 15;   // 128-col slice
        const int lq = lq0 + r;
        const int* mrowp = mask + ((size_t)bb * LQ + lq) * LK;
        float* srow = sc + (size_t)r * LK;

        float mx = -3.0e38f;
        for (int i = 0; i < 128; ++i) {
            int c = j * 128 + i;
            float s = srow[c];
            if (mrowp[c] == 0) { s = -1.0e9f; srow[c] = s; }
            mx = fmaxf(mx, s);
        }
#pragma unroll
        for (int m = 1; m < 16; m <<= 1) mx = fmaxf(mx, __shfl_xor(mx, m, 32));

        float sum = 0.0f;
        for (int i = 0; i < 128; ++i) {
            int c = j * 128 + i;
            float p = __expf(srow[c] - mx);
            srow[c] = p;
            sum += p;
        }
#pragma unroll
        for (int m = 1; m < 16; m <<= 1) sum += __shfl_xor(sum, m, 32);
        const float inv = 1.0f / sum;

        float* arow = attnOut + (((size_t)bb * NHEAD + hh) * LQ + lq) * (size_t)LK;
        for (int i = 0; i < 32; ++i) {
            int c = j * 128 + i * 4;
            f4v p = *(f4v*)&srow[c];
            p *= inv;
            *(f4v*)&srow[c] = p;   // keep normalized probs for PV
            *(f4v*)&arow[c] = p;   // materialize attn output
        }
    }
    __syncthreads();

    // ---- phase 3: out = P V  (waves: 4 d-tiles x 2 K-halves) ----
    const int dt = wave & 3, khalf = wave >> 2;
    const int d0 = dt * 16;
    const _Float16* vbase =
        vhT + (((size_t)bb * NHEAD + hh) * DK + d0 + mrow) * LK + khalf * 1024;
    const float* pbase = sc + (size_t)mrow * LK + khalf * 1024;

    v8f acc = {};
    for (int kt = 0; kt < 32; ++kt) {
        const int kk = kt * 32;
        if ((kt & 7) == 0) __builtin_prefetch(vbase + kk + 256, 0, 0);
        v16h a = load_frag_f32(pbase + kk, kb, 1.0f);   // probs f32->f16 (LDS)
        v16h b = load_frag_f16(vbase + kk, kb);
        acc = wmma16(a, b, acc);
    }
    if (khalf == 1) {
#pragma unroll
        for (int v = 0; v < 8; ++v) red[dt * 256 + (kb + v) * 16 + mrow] = acc[v];
    }
    __syncthreads();
    if (khalf == 0) {
        _Float16* cbase = ctx + ((size_t)bb * LQ + lq0) * (NHEAD * DV) +
                          hh * DV + d0 + mrow;
#pragma unroll
        for (int v = 0; v < 8; ++v) {
            float o = acc[v] + red[dt * 256 + (kb + v) * 16 + mrow];
            cbase[(size_t)(kb + v) * (NHEAD * DV)] = (_Float16)o;
        }
    }
}

// ---------------- K3: out = ctx @ Wfc, concat residual, layernorm ----------------
__global__ __launch_bounds__(256) void fcln_kernel(const _Float16* __restrict__ ctx,
                                                   const _Float16* __restrict__ WfcT,
                                                   const float* __restrict__ qres,
                                                   const float* __restrict__ gamma,
                                                   const float* __restrict__ beta,
                                                   float* __restrict__ out) {
    __shared__ float buf[16 * 768];  // 48KB: [16 rows][512 fc + 256 residual]
    const int lane = threadIdx.x & 31;
    const int wave = threadIdx.x >> 5;
    const int lq0  = blockIdx.x * 16;
    const int bb   = blockIdx.y;
    const int mrow = lane & 15;
    const int kb   = (lane >> 4) * 8;

    const _Float16* crow = ctx + ((size_t)bb * LQ + lq0 + mrow) * DMODEL;
    const int n0 = wave * 64;

    v8f acc[4] = {};
    for (int kk = 0; kk < DMODEL; kk += 32) {
        v16h a = load_frag_f16(crow + kk, kb);   // shared across the 4 n-tiles
#pragma unroll
        for (int t = 0; t < 4; ++t) {
            const _Float16* wrow = WfcT + (size_t)(n0 + t * 16 + mrow) * DMODEL + kk;
            acc[t] = wmma16(a, load_frag_f16(wrow, kb), acc[t]);
        }
    }
#pragma unroll
    for (int t = 0; t < 4; ++t)
#pragma unroll
        for (int v = 0; v < 8; ++v)
            buf[(kb + v) * 768 + n0 + t * 16 + mrow] = acc[t][v];

    {   // stage residual q rows next to fc result
        const int r = threadIdx.x >> 4, j = threadIdx.x & 15;
        const float* qrow = qres + ((size_t)bb * LQ + lq0 + r) * DQRY;
#pragma unroll
        for (int i = 0; i < 4; ++i)
            *(f4v*)&buf[r * 768 + DMODEL + j * 16 + i * 4] =
                *(const f4v*)&qrow[j * 16 + i * 4];
    }
    __syncthreads();

    const int r = threadIdx.x >> 4, j = threadIdx.x & 15;
    float sum = 0.0f, sq = 0.0f;
    for (int i = 0; i < 48; ++i) {
        float x = buf[r * 768 + j * 48 + i];
        sum += x; sq += x * x;
    }
#pragma unroll
    for (int m = 1; m < 16; m <<= 1) {
        sum += __shfl_xor(sum, m, 32);
        sq  += __shfl_xor(sq, m, 32);
    }
    const float mu   = sum * (1.0f / 768.0f);
    const float var  = sq * (1.0f / 768.0f) - mu * mu;
    const float rstd = rsqrtf(var + 1.0e-6f);

    float* orow = out + ((size_t)bb * LQ + lq0 + r) * 768;
    for (int i = 0; i < 48; ++i) {
        int c = j * 48 + i;
        orow[c] = (buf[r * 768 + c] - mu) * rstd * gamma[c] + beta[c];
    }
}

// ---------------- host: launch pipeline ----------------
extern "C" void kernel_launch(void* const* d_in, const int* in_sizes, int n_in,
                              void* d_out, int out_size, void* d_ws, size_t ws_size,
                              hipStream_t stream) {
    const float* q    = (const float*)d_in[0];
    const float* k    = (const float*)d_in[1];
    const float* v    = (const float*)d_in[2];
    const int*   mask = (const int*)d_in[3];
    const float* Wq   = (const float*)d_in[4];
    const float* Wk   = (const float*)d_in[5];
    const float* Wv   = (const float*)d_in[6];
    const float* Wfc  = (const float*)d_in[7];
    const float* gamma = (const float*)d_in[8];
    const float* beta  = (const float*)d_in[9];

    float* out  = (float*)d_out;                          // [B][LQ][768]
    float* attn = out + (size_t)BATCH * LQ * 768;         // [B][H][LQ][LK]

    // workspace carve-up (f16 scratch, ~18 MB total)
    char* ws = (char*)d_ws;
    _Float16* WqT  = (_Float16*)ws;  ws += (size_t)DMODEL * DQRY   * 2;
    _Float16* WkT  = (_Float16*)ws;  ws += (size_t)DMODEL * DMODEL * 2;
    _Float16* WvT  = (_Float16*)ws;  ws += (size_t)DMODEL * DMODEL * 2;
    _Float16* WfcT = (_Float16*)ws;  ws += (size_t)DMODEL * DMODEL * 2;
    _Float16* qh   = (_Float16*)ws;  ws += (size_t)BATCH * NHEAD * LQ * DK * 2;
    _Float16* kh   = (_Float16*)ws;  ws += (size_t)BATCH * NHEAD * LK * DK * 2;
    _Float16* vhT  = (_Float16*)ws;  ws += (size_t)BATCH * NHEAD * DK * LK * 2;
    _Float16* ctx  = (_Float16*)ws;  ws += (size_t)BATCH * LQ * DMODEL * 2;

    // K0: weights -> f16 transposed
    wtrans_kernel<<<(DQRY * DMODEL + 255) / 256, 256, 0, stream>>>(Wq, WqT, DQRY, DMODEL);
    wtrans_kernel<<<(DMODEL * DMODEL + 255) / 256, 256, 0, stream>>>(Wk, WkT, DMODEL, DMODEL);
    wtrans_kernel<<<(DMODEL * DMODEL + 255) / 256, 256, 0, stream>>>(Wv, WvT, DMODEL, DMODEL);
    wtrans_kernel<<<(DMODEL * DMODEL + 255) / 256, 256, 0, stream>>>(Wfc, WfcT, DMODEL, DMODEL);

    // K1: projections (rows = B*L = 4096 -> 256 row-blocks; 512 cols -> 4 col-blocks)
    proj_kernel<<<dim3(256, 4), 256, 0, stream>>>(q, WqT, qh, DQRY, 0);
    proj_kernel<<<dim3(256, 4), 256, 0, stream>>>(k, WkT, kh, DMODEL, 1);
    proj_kernel<<<dim3(256, 4), 256, 0, stream>>>(v, WvT, vhT, DMODEL, 2);

    // K2: attention strip kernel; LDS = 128KB scores + 4KB reduce + 32KB TDM K-tiles
    const size_t attn_lds = (size_t)16 * LK * 4 + 4 * 16 * 16 * 4 + 8 * 2 * 2048;
    (void)hipFuncSetAttribute((const void*)attn_kernel,
                              hipFuncAttributeMaxDynamicSharedMemorySize,
                              (int)attn_lds);
    attn_kernel<<<dim3(LQ / 16, NHEAD, BATCH), 256, attn_lds, stream>>>(
        qh, kh, vhT, mask, attn, ctx);

    // K3: fc + residual concat + layernorm
    fcln_kernel<<<dim3(LQ / 16, BATCH), 256, 0, stream>>>(ctx, WfcT, q, gamma, beta, out);
}
